// PointTransformerSeg_68281390072572
// MI455X (gfx1250) — matrix operand order repaired and checked by reading, not compiled, not run on hardware
//
#include <hip/hip_runtime.h>
#include <cstdint>
#include <cstddef>

#define BATCH 2
#define NPTS  8192
#define CH    64
#define KNB   32
#define SA    80   // f16 LDS row stride (halves) for activation tiles
#define SF    68   // f32 LDS row stride (floats)

typedef _Float16 v16h __attribute__((ext_vector_type(16)));
typedef float    v8f  __attribute__((ext_vector_type(8)));

// ---------------------------------------------------------------- WMMA helpers
static __device__ inline v8f wmma16(v16h a, v16h b, v8f c) {
  // D = A(16x32 f16) * B(32x16 f16) + C(16x16 f32)
  return __builtin_amdgcn_wmma_f32_16x16x32_f16(false, a, false, b, (short)0, c,
                                                false, false);
}

// A fragment from LDS (row-major f16, stride SA halves), tile origin (row0,k0).
// ISA 16-bit A 16x32 layout: lanes 0-15 row M=lane; VGPR v holds
// K = (v<4?0:16)+(v&3)*2 (+8 for lanes 16-31).
static __device__ inline v16h load_A(const _Float16* s, int row0, int k0, int lane) {
  const int row = row0 + (lane & 15);
  const int hi8 = (lane & 16) ? 8 : 0;
  const _Float16* base = s + row * SA + k0 + hi8;
  v16h a;
#pragma unroll
  for (int v = 0; v < 8; ++v) {
    const int kloc = ((v < 4) ? 0 : 16) + (v & 3) * 2;
    a[2 * v]     = base[kloc];
    a[2 * v + 1] = base[kloc + 1];
  }
  return a;
}

// B fragment from pre-swizzled global weights: 16 contiguous halves per lane.
static __device__ inline v16h load_B(const _Float16* w, int kk, int nt, int lane) {
  return *(const v16h*)(w + ((((kk * 4 + nt) * 32) + lane) << 4));
}

// 64-thread block sum
static __device__ inline float redsum64(float v, volatile float* red, int c) {
  red[c] = v; __syncthreads();
  if (c < 32) red[c] += red[c + 32]; __syncthreads();
  if (c < 16) red[c] += red[c + 16]; __syncthreads();
  if (c <  8) red[c] += red[c +  8]; __syncthreads();
  if (c <  4) red[c] += red[c +  4]; __syncthreads();
  if (c <  2) red[c] += red[c +  2]; __syncthreads();
  if (c <  1) red[c] += red[c +  1]; __syncthreads();
  float s = red[0]; __syncthreads();
  return s;
}

// ------------------------------------------------- 1) weight swizzle to f16 frags
// wswz[mat][kk][nt][lane][e] = W[kk*32 + (lane&16?8:0) + (e<8?e:e+8)][nt*16 + (lane&15)]
__global__ __launch_bounds__(256) void k_prep_w(const float* __restrict__ Wk,
                                                const float* __restrict__ Wv,
                                                const float* __restrict__ dW2,
                                                const float* __restrict__ gW1,
                                                const float* __restrict__ gW2,
                                                _Float16* __restrict__ wswz) {
  const int bid = blockIdx.x;
  const float* W = (bid == 0) ? Wk : (bid == 1) ? Wv : (bid == 2) ? dW2
                   : (bid == 3) ? gW1 : gW2;
  _Float16* dst = wswz + bid * 4096;
  for (int f = threadIdx.x; f < 4096; f += 256) {
    const int e    = f & 15;
    const int lane = (f >> 4) & 31;
    const int nt   = (f >> 9) & 3;
    const int kk   = f >> 11;
    const int K = kk * 32 + ((lane & 16) ? 8 : 0) + ((e < 8) ? e : e + 8);
    const int n = nt * 16 + (lane & 15);
    dst[f] = (_Float16)W[K * CH + n];
  }
}

// ------------------------------------------------- 2) fn = LN1(features); q = fn@Wq+bq
__global__ __launch_bounds__(64) void k_prep_fnq(const float* __restrict__ features,
                                                 const float* __restrict__ g,
                                                 const float* __restrict__ b,
                                                 const float* __restrict__ Wq,
                                                 const float* __restrict__ bq,
                                                 _Float16* __restrict__ fnh,
                                                 float* __restrict__ qbuf) {
  __shared__ float srow[64];
  __shared__ float red[64];
  const int row = blockIdx.x, c = threadIdx.x;
  const float x = features[(size_t)row * CH + c];
  const float m = redsum64(x, red, c) * (1.f / CH);
  const float d = x - m;
  const float var = redsum64(d * d, red, c) * (1.f / CH);
  const float fn = d * rsqrtf(var + 1e-5f) * g[c] + b[c];
  srow[c] = fn;
  fnh[(size_t)row * CH + c] = (_Float16)fn;
  __syncthreads();
  float acc = bq[c];
#pragma unroll 8
  for (int i = 0; i < CH; ++i) acc += srow[i] * Wq[i * CH + c];
  qbuf[(size_t)row * CH + c] = acc;
}

// ------------------------------------------------- 3) exact kNN (top-32 by sq dist)
__global__ __launch_bounds__(128) void k_knn(const float* __restrict__ xyz,
                                             int* __restrict__ knn) {
  __shared__ float cpt[3][128];
  __shared__ float sd[KNB * 128];
  __shared__ int   si[KNB * 128];
  const int t = threadIdx.x;
  const int b = blockIdx.x >> 6;      // NPTS/128 = 64 tiles per batch
  const int qn = (blockIdx.x & 63) * 128 + t;
  const size_t qo = ((size_t)b * NPTS + qn) * 3;
  const float qx = xyz[qo], qy = xyz[qo + 1], qz = xyz[qo + 2];
#pragma unroll
  for (int s = 0; s < KNB; ++s) { sd[s * 128 + t] = 3.4e38f; si[s * 128 + t] = 0; }
  for (int base = 0; base < NPTS; base += 128) {
    __syncthreads();
    const size_t co = ((size_t)b * NPTS + base + t) * 3;
    cpt[0][t] = xyz[co]; cpt[1][t] = xyz[co + 1]; cpt[2][t] = xyz[co + 2];
    __syncthreads();
    for (int j = 0; j < 128; ++j) {
      const float dx = qx - cpt[0][j], dy = qy - cpt[1][j], dz = qz - cpt[2][j];
      const float d = dx * dx + dy * dy + dz * dz;
      if (d < sd[(KNB - 1) * 128 + t]) {
        int pos = KNB - 1;
        while (pos > 0 && sd[(pos - 1) * 128 + t] > d) {
          sd[pos * 128 + t] = sd[(pos - 1) * 128 + t];
          si[pos * 128 + t] = si[(pos - 1) * 128 + t];
          --pos;
        }
        sd[pos * 128 + t] = d;
        si[pos * 128 + t] = base + j;
      }
    }
  }
  for (int s = 0; s < KNB; ++s)
    knn[((size_t)b * NPTS + qn) * KNB + s] = si[s * 128 + t];
}

// ------------------------------------------------- 4) fused neighborhood attention
// one wave = one query point; 32 neighbor rows x 64 ch; all 64x64 GEMMs via WMMA
__global__ __launch_bounds__(64) void k_attn(
    const float* __restrict__ xyz, const float* __restrict__ qbuf,
    const _Float16* __restrict__ fnh, const int* __restrict__ knn,
    const _Float16* __restrict__ wswz,
    const float* __restrict__ bk, const float* __restrict__ bv,
    const float* __restrict__ dW1, const float* __restrict__ db1,
    const float* __restrict__ dg, const float* __restrict__ dbeta,
    const float* __restrict__ db2,
    const float* __restrict__ gb1, const float* __restrict__ gg,
    const float* __restrict__ gbeta,
    float* __restrict__ attn_out) {
  __shared__ _Float16 s_act[2][32 * SA];
  __shared__ float    s_tmp[2][32 * SF];
  __shared__ float    s_vpe[2][32 * SF];
  const int wave = threadIdx.x >> 5;
  const int lane = threadIdx.x & 31;
  const int p = blockIdx.x * 2 + wave;     // global point id
  const int b = p / NPTS;
  _Float16* act = s_act[wave];
  float* tmp = s_tmp[wave];
  float* vpe = s_vpe[wave];
  _Float16* hbuf = (_Float16*)tmp;         // h1 staged as f16 (stride SA)

  const int nbr = knn[(size_t)p * KNB + lane];   // lane <-> neighbor row

  // --- fc_delta hidden: h1 = relu(LN(pos_diff @ dW1 + db1)) -> hbuf (f16)
  {
    const float px = xyz[(size_t)p * 3], py = xyz[(size_t)p * 3 + 1],
                pz = xyz[(size_t)p * 3 + 2];
    const size_t no = ((size_t)b * NPTS + nbr) * 3;
    const float dx = px - xyz[no], dy = py - xyz[no + 1], dz = pz - xyz[no + 2];
    float h[CH];
    float s = 0.f;
#pragma unroll
    for (int c = 0; c < CH; ++c) {
      h[c] = dx * dW1[c] + dy * dW1[CH + c] + dz * dW1[2 * CH + c] + db1[c];
      s += h[c];
    }
    const float m = s * (1.f / CH);
    float v = 0.f;
#pragma unroll
    for (int c = 0; c < CH; ++c) { const float dd = h[c] - m; v += dd * dd; }
    const float rs = rsqrtf(v * (1.f / CH) + 1e-5f);
#pragma unroll
    for (int c = 0; c < CH; ++c)
      hbuf[lane * SA + c] = (_Float16)fmaxf((h[c] - m) * rs * dg[c] + dbeta[c], 0.f);
  }

  // --- gather neighbor features (f16, L2-resident) -> act
  {
    const float4* src = (const float4*)(fnh + ((size_t)b * NPTS + nbr) * CH);
    float4* dst = (float4*)(act + lane * SA);
#pragma unroll
    for (int j = 0; j < 8; ++j) dst[j] = src[j];
  }
  __syncthreads();

  // --- k/v/pe GEMMs: [32x64] @ [64x64] each, 2x4x2 WMMA tiles
  const _Float16* wk  = wswz;
  const _Float16* wv  = wswz + 4096;
  const _Float16* wd2 = wswz + 2 * 4096;
  const _Float16* wg1 = wswz + 3 * 4096;
  const _Float16* wg2 = wswz + 4 * 4096;
  v8f kacc[2][4], vacc[2][4], peacc[2][4];
  const v8f z = {};
#pragma unroll
  for (int mt = 0; mt < 2; ++mt)
#pragma unroll
    for (int nt = 0; nt < 4; ++nt) { kacc[mt][nt] = z; vacc[mt][nt] = z; peacc[mt][nt] = z; }
#pragma unroll
  for (int kk = 0; kk < 2; ++kk)
#pragma unroll
    for (int mt = 0; mt < 2; ++mt) {
      const v16h a  = load_A(act,  mt * 16, kk * 32, lane);
      const v16h ah = load_A(hbuf, mt * 16, kk * 32, lane);
#pragma unroll
      for (int nt = 0; nt < 4; ++nt) {
        kacc[mt][nt]  = wmma16(a,  load_B(wk,  kk, nt, lane), kacc[mt][nt]);
        vacc[mt][nt]  = wmma16(a,  load_B(wv,  kk, nt, lane), vacc[mt][nt]);
        peacc[mt][nt] = wmma16(ah, load_B(wd2, kk, nt, lane), peacc[mt][nt]);
      }
    }
  __syncthreads();

  // --- combine: attn_in = q - (k+bk) + (pe+db2) -> act(f16); v+bv+pe -> vpe(f32)
  const int hi8 = (lane & 16) ? 8 : 0;
#pragma unroll
  for (int mt = 0; mt < 2; ++mt)
#pragma unroll
    for (int nt = 0; nt < 4; ++nt) {
      const int n = nt * 16 + (lane & 15);
      const float qn = qbuf[(size_t)p * CH + n];
      const float bkn = bk[n], bvn = bv[n], b2n = db2[n];
#pragma unroll
      for (int v = 0; v < 8; ++v) {
        const int m = mt * 16 + hi8 + v;
        const float pe = peacc[mt][nt][v] + b2n;
        act[m * SA + n] = (_Float16)(qn - (kacc[mt][nt][v] + bkn) + pe);
        vpe[m * SF + n] = vacc[mt][nt][v] + bvn + pe;
      }
    }
  __syncthreads();

  // --- g1 = attn_in @ gW1 + gb1 -> tmp (f32)
  v8f g1[2][4];
#pragma unroll
  for (int mt = 0; mt < 2; ++mt)
#pragma unroll
    for (int nt = 0; nt < 4; ++nt) g1[mt][nt] = z;
#pragma unroll
  for (int kk = 0; kk < 2; ++kk)
#pragma unroll
    for (int mt = 0; mt < 2; ++mt) {
      const v16h a = load_A(act, mt * 16, kk * 32, lane);
#pragma unroll
      for (int nt = 0; nt < 4; ++nt)
        g1[mt][nt] = wmma16(a, load_B(wg1, kk, nt, lane), g1[mt][nt]);
    }
#pragma unroll
  for (int mt = 0; mt < 2; ++mt)
#pragma unroll
    for (int nt = 0; nt < 4; ++nt) {
      const int n = nt * 16 + (lane & 15);
      const float bb = gb1[n];
#pragma unroll
      for (int v = 0; v < 8; ++v)
        tmp[(mt * 16 + hi8 + v) * SF + n] = g1[mt][nt][v] + bb;
    }
  __syncthreads();

  // --- per-row LN + relu over 64 ch (row = neighbor = lane) -> act (f16)
  {
    float r[CH];
    float s = 0.f;
#pragma unroll
    for (int c = 0; c < CH; ++c) { r[c] = tmp[lane * SF + c]; s += r[c]; }
    const float m = s * (1.f / CH);
    float v = 0.f;
#pragma unroll
    for (int c = 0; c < CH; ++c) { const float dd = r[c] - m; v += dd * dd; }
    const float rs = rsqrtf(v * (1.f / CH) + 1e-5f);
#pragma unroll
    for (int c = 0; c < CH; ++c)
      act[lane * SA + c] = (_Float16)fmaxf((r[c] - m) * rs * gg[c] + gbeta[c], 0.f);
  }
  __syncthreads();

  // --- a = h2 @ gW2 -> tmp (f32). gb2 constant along K => cancels in softmax.
  v8f aa[2][4];
#pragma unroll
  for (int mt = 0; mt < 2; ++mt)
#pragma unroll
    for (int nt = 0; nt < 4; ++nt) aa[mt][nt] = z;
#pragma unroll
  for (int kk = 0; kk < 2; ++kk)
#pragma unroll
    for (int mt = 0; mt < 2; ++mt) {
      const v16h a = load_A(act, mt * 16, kk * 32, lane);
#pragma unroll
      for (int nt = 0; nt < 4; ++nt)
        aa[mt][nt] = wmma16(a, load_B(wg2, kk, nt, lane), aa[mt][nt]);
    }
#pragma unroll
  for (int mt = 0; mt < 2; ++mt)
#pragma unroll
    for (int nt = 0; nt < 4; ++nt) {
      const int n = nt * 16 + (lane & 15);
#pragma unroll
      for (int v = 0; v < 8; ++v)
        tmp[(mt * 16 + hi8 + v) * SF + n] = aa[mt][nt][v];
    }
  __syncthreads();

  // --- softmax over the 32 neighbors per channel + weighted sum with v+pe
  const float scale = 0.35355339059327373f;   // 1/sqrt(HEAD_DIM=8)
#pragma unroll
  for (int t = 0; t < 2; ++t) {
    const int c = lane + 32 * t;
    float mx = -3.4e38f;
    for (int k2 = 0; k2 < KNB; ++k2) mx = fmaxf(mx, tmp[k2 * SF + c]);
    float se = 0.f, o = 0.f;
    for (int k2 = 0; k2 < KNB; ++k2) {
      const float e = __expf((tmp[k2 * SF + c] - mx) * scale);
      se += e;
      o += e * vpe[k2 * SF + c];
    }
    attn_out[(size_t)p * CH + c] = o / se;
  }
}

// ------------------------------------------------- 5) out = LN2(out@Wo + bo), in place
__global__ __launch_bounds__(64) void k_fcout(float* __restrict__ buf,
                                              const float* __restrict__ Wo,
                                              const float* __restrict__ bo,
                                              const float* __restrict__ g,
                                              const float* __restrict__ b) {
  __shared__ float srow[64];
  __shared__ float red[64];
  const int row = blockIdx.x, c = threadIdx.x;
  srow[c] = buf[(size_t)row * CH + c];
  __syncthreads();
  float acc = bo[c];
#pragma unroll 8
  for (int i = 0; i < CH; ++i) acc += srow[i] * Wo[i * CH + c];
  const float m = redsum64(acc, red, c) * (1.f / CH);
  const float d = acc - m;
  const float var = redsum64(d * d, red, c) * (1.f / CH);
  buf[(size_t)row * CH + c] = d * rsqrtf(var + 1e-5f) * g[c] + b[c];
}

// ------------------------------------------------- 6) channel attention weights
__global__ __launch_bounds__(64) void k_chan(const float* __restrict__ buf,
                                             const float* __restrict__ caW1,
                                             const float* __restrict__ cab1,
                                             const float* __restrict__ caW2,
                                             const float* __restrict__ cab2,
                                             float* __restrict__ cw) {
  __shared__ float mrow[64];
  __shared__ float hid[16];
  const int b = blockIdx.x, c = threadIdx.x;
  float s = 0.f;
  const float* p = buf + (size_t)b * NPTS * CH + c;
  for (int n = 0; n < NPTS; ++n) s += p[(size_t)n * CH];
  mrow[c] = s * (1.f / NPTS);
  __syncthreads();
  if (c < 16) {
    float h = cab1[c];
#pragma unroll
    for (int i = 0; i < CH; ++i) h += mrow[i] * caW1[i * 16 + c];
    hid[c] = fmaxf(h, 0.f);
  }
  __syncthreads();
  float o = cab2[c];
#pragma unroll
  for (int i = 0; i < 16; ++i) o += hid[i] * caW2[i * CH + c];
  cw[b * CH + c] = 1.f / (1.f + __expf(-o));
}

// ------------------------------------------------- 7) LN3(out*cw) + features
__global__ __launch_bounds__(64) void k_final(const float* __restrict__ buf,
                                              const float* __restrict__ cw,
                                              const float* __restrict__ features,
                                              const float* __restrict__ g,
                                              const float* __restrict__ b,
                                              float* __restrict__ out) {
  __shared__ float red[64];
  const int row = blockIdx.x, c = threadIdx.x;
  const int bidx = row / NPTS;
  const float x = buf[(size_t)row * CH + c] * cw[bidx * CH + c];
  const float m = redsum64(x, red, c) * (1.f / CH);
  const float d = x - m;
  const float var = redsum64(d * d, red, c) * (1.f / CH);
  out[(size_t)row * CH + c] =
      d * rsqrtf(var + 1e-5f) * g[c] + b[c] + features[(size_t)row * CH + c];
}

// ------------------------------------------------------------------- launcher
extern "C" void kernel_launch(void* const* d_in, const int* in_sizes, int n_in,
                              void* d_out, int out_size, void* d_ws, size_t ws_size,
                              hipStream_t stream) {
  (void)in_sizes; (void)n_in; (void)out_size; (void)ws_size;
  const float* xyz      = (const float*)d_in[0];
  const float* features = (const float*)d_in[1];
  const float* ln1g = (const float*)d_in[2];
  const float* ln1b = (const float*)d_in[3];
  const float* Wq = (const float*)d_in[4];
  const float* bq = (const float*)d_in[5];
  const float* Wk = (const float*)d_in[6];
  const float* bk = (const float*)d_in[7];
  const float* Wv = (const float*)d_in[8];
  const float* bv = (const float*)d_in[9];
  const float* dW1 = (const float*)d_in[10];
  const float* db1 = (const float*)d_in[11];
  const float* dg  = (const float*)d_in[12];
  const float* dbt = (const float*)d_in[13];
  const float* dW2 = (const float*)d_in[14];
  const float* db2 = (const float*)d_in[15];
  const float* gW1 = (const float*)d_in[16];
  const float* gb1 = (const float*)d_in[17];
  const float* gg  = (const float*)d_in[18];
  const float* gbt = (const float*)d_in[19];
  const float* gW2 = (const float*)d_in[20];
  const float* Wo  = (const float*)d_in[22];
  const float* bo  = (const float*)d_in[23];
  const float* ln2g = (const float*)d_in[24];
  const float* ln2b = (const float*)d_in[25];
  const float* caW1 = (const float*)d_in[26];
  const float* cab1 = (const float*)d_in[27];
  const float* caW2 = (const float*)d_in[28];
  const float* cab2 = (const float*)d_in[29];
  const float* ln3g = (const float*)d_in[30];
  const float* ln3b = (const float*)d_in[31];
  float* out = (float*)d_out;

  char* ws = (char*)d_ws;
  size_t off = 0;
  auto carve = [&](size_t bytes) {
    char* p = ws + off;
    off += (bytes + 255) & ~(size_t)255;
    return p;
  };
  const size_t ROWS = (size_t)BATCH * NPTS;
  _Float16* fnh  = (_Float16*)carve(ROWS * CH * sizeof(_Float16));
  float*    qbuf = (float*)carve(ROWS * CH * sizeof(float));
  int*      knn  = (int*)carve(ROWS * KNB * sizeof(int));
  _Float16* wswz = (_Float16*)carve(5 * 4096 * sizeof(_Float16));
  float*    abuf = (float*)carve(ROWS * CH * sizeof(float));   // attn out, then LN2 out in-place
  float*    cw   = (float*)carve(BATCH * CH * sizeof(float));

  k_prep_w  <<<5, 256, 0, stream>>>(Wk, Wv, dW2, gW1, gW2, wswz);
  k_prep_fnq<<<(int)ROWS, 64, 0, stream>>>(features, ln1g, ln1b, Wq, bq, fnh, qbuf);
  k_knn     <<<BATCH * (NPTS / 128), 128, 0, stream>>>(xyz, knn);
  k_attn    <<<(int)(ROWS / 2), 64, 0, stream>>>(xyz, qbuf, fnh, knn, wswz,
                                                 bk, bv, dW1, db1, dg, dbt, db2,
                                                 gb1, gg, gbt, abuf);
  k_fcout   <<<(int)ROWS, 64, 0, stream>>>(abuf, Wo, bo, ln2g, ln2b);
  k_chan    <<<BATCH, 64, 0, stream>>>(abuf, caW1, cab1, caW2, cab2, cw);
  k_final   <<<(int)ROWS, 64, 0, stream>>>(abuf, cw, features, ln3g, ln3b, out);
}
